// MemoryBlock_86208583565576
// MI455X (gfx1250) — compile-verified, matrix operand order checked by reading, not run on hardware
//
#include <hip/hip_runtime.h>
#include <hip/hip_bf16.h>
#include <cstdint>

typedef __attribute__((ext_vector_type(16))) __bf16 v16bf;
typedef __attribute__((ext_vector_type(8)))  __bf16 v8bf;
typedef __attribute__((ext_vector_type(8)))  float  v8f;

#define EPSF 1e-6f

// Problem dims (fixed by the reference)
constexpr int BATCH = 8;
constexpr int KS    = 64;     // slots
constexpr int MS    = 4096;   // tokens
constexpr int D     = 1024;
constexpr int D3    = 3072;

// ---------------------------------------------------------------------------
// CDNA5 async copy helpers (inline asm: portable across ROCm7.2 / clang-23)
// ---------------------------------------------------------------------------
__device__ __forceinline__ void async_b128x2(uint32_t lds_byte, const void* gaddr) {
    // copies 32 contiguous bytes global -> LDS (per lane), tracked by ASYNCcnt
    asm volatile("global_load_async_to_lds_b128 %0, %1, off\n\t"
                 "global_load_async_to_lds_b128 %0, %1, off offset:16"
                 :: "v"(lds_byte), "v"(gaddr) : "memory");
}
__device__ __forceinline__ void async_b128(uint32_t lds_byte, const void* gaddr) {
    asm volatile("global_load_async_to_lds_b128 %0, %1, off"
                 :: "v"(lds_byte), "v"(gaddr) : "memory");
}
__device__ __forceinline__ void wait_async0() {
    asm volatile("s_wait_asynccnt 0x0" ::: "memory");
}

// ---------------------------------------------------------------------------
// Generic batched WMMA GEMM:  C[M,N] = scale * A[M,K] * B[N,K]^T   (bf16 in)
// STORE_MODE: 0 -> f32 C[m*N+n], 1 -> bf16 C[m*N+n], 2 -> bf16 C[n*M+m]
// MT: M-tile (128 for big GEMMs, 64 for the slot-sized GEMMs). N-tile = 128.
// 256 threads (8 waves); K staged 32-deep through double-buffered LDS via
// async-to-LDS copies.
// ---------------------------------------------------------------------------
template<int STORE_MODE, int MT>
__global__ __launch_bounds__(256) void gemm_bt_wmma(
    const __bf16* __restrict__ A, const __bf16* __restrict__ Bmat,
    float* __restrict__ Cf, __bf16* __restrict__ Cb,
    int M, int N, int K, long sA, long sB, long sC, float scale)
{
    constexpr int AST   = 40;                 // LDS row stride (elems): 80B, 16B-aligned
    constexpr int WM    = (MT == 128) ? 4 : 2;
    constexpr int WN    = 8 / WM;
    constexpr int TM    = (MT / 16) / WM;     // 2
    constexpr int TN    = (128 / 16) / WN;    // 4 (MT=128) or 2 (MT=64)
    constexpr int ATILE = MT * AST;           // elems
    constexpr int BTILE = 128 * AST;
    __shared__ __bf16 smem[2 * (ATILE + BTILE)];

    const int tid   = threadIdx.x;
    const int lane  = tid & 31;
    const int w     = tid >> 5;
    const int waveM = (MT == 128) ? (w >> 1) : (w >> 2);
    const int waveN = (MT == 128) ? (w & 1)  : (w & 3);
    const int half  = lane >> 4;              // 0/1
    const int lrow  = lane & 15;
    const int kofs  = half * 8;               // K phase for fragments

    const int nBase = blockIdx.x * 128;
    const int mBase = blockIdx.y * MT;
    const int b     = blockIdx.z;

    const __bf16* Ab = A    + (long)b * sA;
    const __bf16* Bp = Bmat + (long)b * sB;

    v8f acc[TM][TN];
    #pragma unroll
    for (int i = 0; i < TM; ++i)
        #pragma unroll
        for (int j = 0; j < TN; ++j) acc[i][j] = v8f{};

    // staging geometry: A tile MT x 32, B tile 128 x 32 (bf16)
    const int aRow = (MT == 128) ? (tid >> 1) : (tid >> 2);
    const int aCh  = (MT == 128) ? ((tid & 1) * 16) : ((tid & 3) * 8);
    const int bRow = tid >> 1;
    const int bCh  = (tid & 1) * 16;

    const int gAr  = (mBase + aRow < M) ? (mBase + aRow) : (M - 1); // clamp, rows >= M never stored
    const long aOff = (long)gAr * K + aCh;
    const long bOff = (long)(nBase + bRow) * K + bCh;

    const uint32_t laBase = (uint32_t)(uintptr_t)&smem[aRow * AST + aCh];
    const uint32_t lbBase = (uint32_t)(uintptr_t)&smem[ATILE + bRow * AST + bCh];
    constexpr uint32_t BUFB = (uint32_t)(2 * (ATILE + BTILE));     // byte stride between buffers

    auto stage = [&](int buf, int k0) {
        const uint32_t bo = (uint32_t)buf * BUFB;
        if (MT == 128) async_b128x2(laBase + bo, Ab + aOff + k0);
        else           async_b128 (laBase + bo, Ab + aOff + k0);
        async_b128x2(lbBase + bo, Bp + bOff + k0);
    };

    stage(0, 0);
    wait_async0();
    __syncthreads();

    int cur = 0;
    for (int k0 = 0; k0 < K; k0 += 32) {
        if (k0 + 32 < K) stage(cur ^ 1, k0 + 32);   // prefetch next tile into other buffer

        const __bf16* As = smem + cur * (ATILE + BTILE);
        const __bf16* Bs = As + ATILE;

        // fragments (lanes 0-15: K {0..7,16..23}; lanes 16-31: K {8..15,24..31})
        v16bf afrag[TM], bfrag[TN];
        #pragma unroll
        for (int tm = 0; tm < TM; ++tm) {
            const int r = waveM * (TM * 16) + tm * 16 + lrow;
            v8bf lo = *(const v8bf*)&As[r * AST + kofs];
            v8bf hi = *(const v8bf*)&As[r * AST + kofs + 16];
            #pragma unroll
            for (int i = 0; i < 8; ++i) { afrag[tm][i] = lo[i]; afrag[tm][i + 8] = hi[i]; }
        }
        #pragma unroll
        for (int tn = 0; tn < TN; ++tn) {
            const int r = waveN * (TN * 16) + tn * 16 + lrow;
            v8bf lo = *(const v8bf*)&Bs[r * AST + kofs];
            v8bf hi = *(const v8bf*)&Bs[r * AST + kofs + 16];
            #pragma unroll
            for (int i = 0; i < 8; ++i) { bfrag[tn][i] = lo[i]; bfrag[tn][i + 8] = hi[i]; }
        }

        #pragma unroll
        for (int tm = 0; tm < TM; ++tm)
            #pragma unroll
            for (int tn = 0; tn < TN; ++tn)
                acc[tm][tn] = __builtin_amdgcn_wmma_f32_16x16x32_bf16(
                    false, afrag[tm], false, bfrag[tn],
                    (short)0, acc[tm][tn], false, false);

        wait_async0();     // my async writes (next buffer) landed in LDS
        __syncthreads();   // everyone's landed; everyone done reading cur
        cur ^= 1;
    }

    // store (C/D layout: lane n = lrow, rows m = half*8 + v)
    #pragma unroll
    for (int tm = 0; tm < TM; ++tm) {
        const int mrow = mBase + waveM * (TM * 16) + tm * 16 + half * 8;
        #pragma unroll
        for (int tn = 0; tn < TN; ++tn) {
            const int n = nBase + waveN * (TN * 16) + tn * 16 + lrow;
            #pragma unroll
            for (int v = 0; v < 8; ++v) {
                const int m = mrow + v;
                if (m < M) {
                    const float val = acc[tm][tn][v] * scale;
                    if constexpr (STORE_MODE == 0)
                        Cf[(long)b * sC + (long)m * N + n] = val;
                    else if constexpr (STORE_MODE == 1)
                        Cb[(long)b * sC + (long)m * N + n] = (__bf16)val;
                    else
                        Cb[(long)b * sC + (long)n * M + m] = (__bf16)val;
                }
            }
        }
    }
}

// ---------------------------------------------------------------------------
// rmsnorm over last dim (d == 1024), f32 in -> bf16 out. One block per row.
// ---------------------------------------------------------------------------
__global__ __launch_bounds__(256) void rmsnorm_bf16_kernel(
    const float* __restrict__ X, const float* __restrict__ w,
    __bf16* __restrict__ Y)
{
    __shared__ float red[8];
    __shared__ float rs_sh;
    const long row = blockIdx.x;
    const float* x = X + row * D;
    const int tid = threadIdx.x;

    float vals[4];
    float ss = 0.f;
    #pragma unroll
    for (int i = 0; i < 4; ++i) {
        const float v = x[tid + i * 256];
        vals[i] = v; ss += v * v;
    }
    #pragma unroll
    for (int off = 16; off > 0; off >>= 1) ss += __shfl_down(ss, off);
    if ((tid & 31) == 0) red[tid >> 5] = ss;
    __syncthreads();
    if (tid == 0) {
        float t = 0.f;
        #pragma unroll
        for (int i = 0; i < 8; ++i) t += red[i];
        rs_sh = rsqrtf(t * (1.0f / D) + EPSF);
    }
    __syncthreads();
    const float rs = rs_sh;
    #pragma unroll
    for (int i = 0; i < 4; ++i) {
        const int j = tid + i * 256;
        Y[row * D + j] = (__bf16)(vals[i] * rs * w[j]);
    }
}

// ---------------------------------------------------------------------------
__global__ void cast_f32_bf16(const float* __restrict__ X, __bf16* __restrict__ Y, long n)
{
    const long i = blockIdx.x * (long)blockDim.x + threadIdx.x;
    if (i < n) Y[i] = (__bf16)X[i];
}

// scores (B, 64, MS): softmax over the slot axis (dim1), in place.
__global__ __launch_bounds__(256) void softmax_k_kernel(float* __restrict__ S)
{
    const int b = blockIdx.y;
    const int m = blockIdx.x * blockDim.x + threadIdx.x;
    float* s = S + (long)b * KS * MS + m;
    float mx = -3.0e38f;
    #pragma unroll 8
    for (int k = 0; k < KS; ++k) mx = fmaxf(mx, s[(long)k * MS]);
    float sum = 0.f;
    #pragma unroll 8
    for (int k = 0; k < KS; ++k) {
        const float e = __expf(s[(long)k * MS] - mx);
        s[(long)k * MS] = e; sum += e;
    }
    const float inv = 1.0f / sum;
    #pragma unroll 8
    for (int k = 0; k < KS; ++k) s[(long)k * MS] *= inv;
}

// per-(b,k) row sums over m
__global__ __launch_bounds__(256) void rowsum_kernel(const float* __restrict__ S, float* __restrict__ rs)
{
    __shared__ float red[8];
    const int row = blockIdx.x;          // b*KS + k
    const float* s = S + (long)row * MS;
    float t = 0.f;
    for (int j = threadIdx.x; j < MS; j += 256) t += s[j];
    #pragma unroll
    for (int off = 16; off > 0; off >>= 1) t += __shfl_down(t, off);
    if ((threadIdx.x & 31) == 0) red[threadIdx.x >> 5] = t;
    __syncthreads();
    if (threadIdx.x == 0) {
        float a = 0.f;
        #pragma unroll
        for (int i = 0; i < 8; ++i) a += red[i];
        rs[row] = a;
    }
}

__global__ void attn_norm_cast(const float* __restrict__ S, const float* __restrict__ rs,
                               __bf16* __restrict__ Ab, long n)
{
    const long i = blockIdx.x * (long)blockDim.x + threadIdx.x;
    if (i < n) {
        const long row = i / MS;
        Ab[i] = (__bf16)(S[i] / (rs[row] + EPSF));
    }
}

// fused GRU gates: slots <- (1-z)*n + z*slots
__global__ void gru_kernel(const float* __restrict__ gi, const float* __restrict__ gh,
                           const float* __restrict__ bi, const float* __restrict__ bh,
                           float* __restrict__ slots)
{
    const long i = blockIdx.x * (long)blockDim.x + threadIdx.x;   // over BATCH*KS*D
    const long row = i / D;
    const int  j   = (int)(i % D);
    const float* gr = gi + row * (long)D3;
    const float* hr = gh + row * (long)D3;
    const float ir  = gr[j]         + bi[j];
    const float iz  = gr[D + j]     + bi[D + j];
    const float in_ = gr[2 * D + j] + bi[2 * D + j];
    const float hrv = hr[j]         + bh[j];
    const float hz  = hr[D + j]     + bh[D + j];
    const float hn  = hr[2 * D + j] + bh[2 * D + j];
    const float r = 1.f / (1.f + __expf(-(ir + hrv)));
    const float z = 1.f / (1.f + __expf(-(iz + hz)));
    const float n = tanhf(in_ + r * hn);
    const float h = slots[i];
    slots[i] = (1.f - z) * n + z * h;
}

// ---------------------------------------------------------------------------
extern "C" void kernel_launch(void* const* d_in, const int* in_sizes, int n_in,
                              void* d_out, int out_size, void* d_ws, size_t ws_size,
                              hipStream_t stream)
{
    const float* slots_in = (const float*)d_in[0];
    const float* P        = (const float*)d_in[1];
    const float* Wq       = (const float*)d_in[2];
    const float* Wk       = (const float*)d_in[3];
    const float* Wv       = (const float*)d_in[4];
    const float* gwi      = (const float*)d_in[5];
    const float* gwh      = (const float*)d_in[6];
    const float* gbi      = (const float*)d_in[7];
    const float* gbh      = (const float*)d_in[8];
    const float* snw      = (const float*)d_in[9];
    const float* inw      = (const float*)d_in[10];
    float* slots = (float*)d_out;   // working + final slots

    // ---- workspace carve-up (bytes, 256-aligned)
    char* ws = (char*)d_ws;
    size_t off = 0;
    auto alloc = [&](size_t bytes) { void* p = ws + off; off = (off + bytes + 255) & ~(size_t)255; return p; };
    __bf16* Wq_bf   = (__bf16*)alloc((size_t)D * D * 2);
    __bf16* Wk_bf   = (__bf16*)alloc((size_t)D * D * 2);
    __bf16* Wv_bf   = (__bf16*)alloc((size_t)D * D * 2);
    __bf16* wi_bf   = (__bf16*)alloc((size_t)D3 * D * 2);
    __bf16* wh_bf   = (__bf16*)alloc((size_t)D3 * D * 2);
    __bf16* Pn_bf   = (__bf16*)alloc((size_t)BATCH * MS * D * 2);
    __bf16* k_bf    = (__bf16*)alloc((size_t)BATCH * MS * D * 2);
    __bf16* vT_bf   = (__bf16*)alloc((size_t)BATCH * D * MS * 2);   // (d, M) per batch
    __bf16* sn_bf   = (__bf16*)alloc((size_t)BATCH * KS * D * 2);
    __bf16* slot_bf = (__bf16*)alloc((size_t)BATCH * KS * D * 2);
    __bf16* q_bf    = (__bf16*)alloc((size_t)BATCH * KS * D * 2);
    __bf16* upd_bf  = (__bf16*)alloc((size_t)BATCH * KS * D * 2);
    float*  scores  = (float*) alloc((size_t)BATCH * KS * MS * 4);
    float*  rowsum  = (float*) alloc((size_t)BATCH * KS * 4);
    __bf16* attn_bf = (__bf16*)alloc((size_t)BATCH * KS * MS * 2);
    float*  gi      = (float*) alloc((size_t)BATCH * KS * D3 * 4);
    float*  gh      = (float*) alloc((size_t)BATCH * KS * D3 * 4);
    (void)ws_size; (void)in_sizes; (void)n_in; (void)out_size;

    // ---- seed working slots
    hipMemcpyAsync(slots, slots_in, (size_t)BATCH * KS * D * 4, hipMemcpyDeviceToDevice, stream);

    // ---- one-time: weight casts + rmsnorm(P) + k/v projections
    cast_f32_bf16<<<(D * D + 255) / 256, 256, 0, stream>>>(Wq, Wq_bf, (long)D * D);
    cast_f32_bf16<<<(D * D + 255) / 256, 256, 0, stream>>>(Wk, Wk_bf, (long)D * D);
    cast_f32_bf16<<<(D * D + 255) / 256, 256, 0, stream>>>(Wv, Wv_bf, (long)D * D);
    cast_f32_bf16<<<(D3 * D + 255) / 256, 256, 0, stream>>>(gwi, wi_bf, (long)D3 * D);
    cast_f32_bf16<<<(D3 * D + 255) / 256, 256, 0, stream>>>(gwh, wh_bf, (long)D3 * D);

    rmsnorm_bf16_kernel<<<BATCH * MS, 256, 0, stream>>>(P, inw, Pn_bf);

    // k_proj (bf16, row-major (M,d)) and v_proj^T (bf16, (d,M))
    gemm_bt_wmma<1, 128><<<dim3(D / 128, MS / 128, BATCH), 256, 0, stream>>>(
        Pn_bf, Wk_bf, nullptr, k_bf, MS, D, D,
        (long)MS * D, 0, (long)MS * D, 1.0f);
    gemm_bt_wmma<2, 128><<<dim3(D / 128, MS / 128, BATCH), 256, 0, stream>>>(
        Pn_bf, Wv_bf, nullptr, vT_bf, MS, D, D,
        (long)MS * D, 0, (long)D * MS, 1.0f);

    const float scale = 1.0f / 32.0f;   // d^-0.5, d=1024
    for (int it = 0; it < 3; ++it) {
        cast_f32_bf16<<<(BATCH * KS * D + 255) / 256, 256, 0, stream>>>(
            slots, slot_bf, (long)BATCH * KS * D);
        rmsnorm_bf16_kernel<<<BATCH * KS, 256, 0, stream>>>(slots, snw, sn_bf);

        // q = sn @ Wq^T  (bf16 out)
        gemm_bt_wmma<1, 64><<<dim3(D / 128, 1, BATCH), 256, 0, stream>>>(
            sn_bf, Wq_bf, nullptr, q_bf, KS, D, D,
            (long)KS * D, 0, (long)KS * D, 1.0f);

        // scores = scale * q @ k_proj^T  (f32 out)
        gemm_bt_wmma<0, 64><<<dim3(MS / 128, 1, BATCH), 256, 0, stream>>>(
            q_bf, k_bf, scores, nullptr, KS, MS, D,
            (long)KS * D, (long)MS * D, (long)KS * MS, scale);

        softmax_k_kernel<<<dim3(MS / 256, BATCH), 256, 0, stream>>>(scores);
        rowsum_kernel<<<BATCH * KS, 256, 0, stream>>>(scores, rowsum);
        attn_norm_cast<<<(BATCH * KS * MS + 255) / 256, 256, 0, stream>>>(
            scores, rowsum, attn_bf, (long)BATCH * KS * MS);

        // updates = attn @ v_proj = attn @ (vT)^T  (bf16 out)
        gemm_bt_wmma<1, 64><<<dim3(D / 128, 1, BATCH), 256, 0, stream>>>(
            attn_bf, vT_bf, nullptr, upd_bf, KS, D, MS,
            (long)KS * MS, (long)D * MS, (long)KS * D, 1.0f);

        // gi = updates @ gru_wi^T ; gh = slots @ gru_wh^T   (f32, biases folded into gru_kernel)
        gemm_bt_wmma<0, 64><<<dim3(D3 / 128, 1, BATCH), 256, 0, stream>>>(
            upd_bf, wi_bf, gi, nullptr, KS, D3, D,
            (long)KS * D, 0, (long)KS * D3, 1.0f);
        gemm_bt_wmma<0, 64><<<dim3(D3 / 128, 1, BATCH), 256, 0, stream>>>(
            slot_bf, wh_bf, gh, nullptr, KS, D3, D,
            (long)KS * D, 0, (long)KS * D3, 1.0f);

        gru_kernel<<<(BATCH * KS * D + 255) / 256, 256, 0, stream>>>(gi, gh, gbi, gbh, slots);
    }
}
